// Aligner_73607149519242
// MI455X (gfx1250) — matrix operand order repaired and checked by reading, not compiled, not run on hardware
//
#include <hip/hip_runtime.h>
#include <hip/hip_bf16.h>

// ---------------------------------------------------------------------------
// Aligner for MI455X (gfx1250, wave32, WMMA bf16 16x16x32)
//
// Pipeline:
//   k_gate : fused  g = sigmoid(w2 . relu(W1 . relu(x) + b1) + b2)   [B,T]
//            (WMMA bf16 GEMM, 512x512xT per batch, conv2 epilogue in-reg+LDS)
//   k_scan : per-batch cumsum -> centers[B,T], aligned_lengths[B]
//   k_soft : Gaussian-attention softmax -> w[B,L,T] stored bf16
//   k_gemm2: aligned[b,l,c] = sum_t w[b,l,t]*x[b,c,t]  (WMMA bf16 GEMM)
// ---------------------------------------------------------------------------

typedef __attribute__((ext_vector_type(16))) __bf16 v16bf;
typedef __attribute__((ext_vector_type(8)))  float  v8f;

static constexpr int Bn = 32, Cn = 512, Tn = 2048, Ln = 80;

// ---------------------------------------------------------------------------
// Kernel 1: fused gate.  grid = B * (T/16), block = 256 (8 waves).
// Each wave owns 4 o-tiles (16 outputs each): 8 waves * 4 * 16 = 512 = C.
// WMMA: A = conv1_w[o0:o0+16, c:c+32] bf16, B = relu(x[b, c:c+32, t0:t0+16]) bf16.
// Epilogue: p = sum_o relu(acc + b1[o]) * w2[o], reduced over lanes/waves via LDS.
// ---------------------------------------------------------------------------
__global__ __launch_bounds__(256) void k_gate(const float* __restrict__ x,
                                              const float* __restrict__ w1,
                                              const float* __restrict__ b1,
                                              const float* __restrict__ w2,
                                              const float* __restrict__ b2,
                                              float* __restrict__ g)
{
    __shared__ float red[16];

    const int bt   = blockIdx.x;
    const int b    = bt >> 7;           // / (T/16) = /128
    const int t0   = (bt & 127) << 4;   // * 16
    const int lane = threadIdx.x & 31;
    const int wave = threadIdx.x >> 5;
    const int tl   = lane & 15;
    const bool hi  = lane >= 16;

    if (threadIdx.x < 16) red[threadIdx.x] = 0.0f;
    __syncthreads();

    const float* xb = x + (size_t)b * Cn * Tn;
    const int t = t0 + tl;

    v8f acc[4] = {};

    for (int kk = 0; kk < Cn; kk += 32) {
        // ---- B operand: relu(x) tile, 32(K=c) x 16(N=t), bf16 --------------
        // lanes 0-15: K = kk..kk+7, kk+16..kk+23 ; lanes 16-31: K = kk+8..kk+15, kk+24..kk+31
        v16bf bb;
        {
            const float* p = xb + (size_t)(kk + (hi ? 8 : 0)) * Tn + t;
            #pragma unroll
            for (int e = 0; e < 16; ++e) {
                float v = p[(size_t)(e < 8 ? e : e + 8) * Tn];
                bb[e] = (__bf16)(v > 0.0f ? v : 0.0f);
            }
        }
        // ---- 4 A operands + WMMA ------------------------------------------
        #pragma unroll
        for (int j = 0; j < 4; ++j) {
            const int o = ((wave * 4 + j) << 4) + tl;       // A row (M)
            const float* q = w1 + (size_t)o * Cn + kk;
            const int kb = hi ? 8 : 0;
            v16bf aa;
            #pragma unroll
            for (int e = 0; e < 8; ++e) aa[e]     = (__bf16)q[kb + e];       // K=kb..kb+7
            #pragma unroll
            for (int e = 0; e < 8; ++e) aa[8 + e] = (__bf16)q[16 + kb + e];  // K=16+kb..
            acc[j] = __builtin_amdgcn_wmma_f32_16x16x32_bf16(
                         false, aa, false, bb, (short)0, acc[j], false, false);
        }
    }

    // ---- epilogue: conv1 bias + relu + conv2 dot, per-lane partial --------
    float p = 0.0f;
    #pragma unroll
    for (int j = 0; j < 4; ++j) {
        const int obase = ((wave * 4 + j) << 4) + (hi ? 8 : 0);   // M rows this lane holds
        #pragma unroll
        for (int r = 0; r < 8; ++r) {
            const int o = obase + r;
            float v = acc[j][r] + b1[o];
            v = v > 0.0f ? v : 0.0f;
            p += v * w2[o];
        }
    }
    atomicAdd(&red[tl], p);     // ds_add_f32: reduce over 8 waves * 2 half-waves
    __syncthreads();

    if (threadIdx.x < 16) {
        const float s = red[threadIdx.x] + b2[0];
        g[(size_t)b * Tn + t0 + threadIdx.x] = 1.0f / (1.0f + __expf(-s));
    }
}

// ---------------------------------------------------------------------------
// Kernel 2: per-batch inclusive scan of g, emit centers and aligned_lengths.
// grid = B, block = 256, 8 elements/thread.
// ---------------------------------------------------------------------------
__global__ __launch_bounds__(256) void k_scan(const float* __restrict__ g,
                                              const int*   __restrict__ len_fea,
                                              float* __restrict__ centers,
                                              float* __restrict__ lens_out)
{
    __shared__ float sums[256];
    const int b   = blockIdx.x;
    const int tid = threadIdx.x;
    const float* gb = g + (size_t)b * Tn;

    float v[8];
    float s = 0.0f;
    #pragma unroll
    for (int i = 0; i < 8; ++i) { v[i] = gb[tid * 8 + i]; s += v[i]; }
    sums[tid] = s;
    __syncthreads();

    for (int off = 1; off < 256; off <<= 1) {
        const float add = (tid >= off) ? sums[tid - off] : 0.0f;
        __syncthreads();
        sums[tid] += add;
        __syncthreads();
    }
    const float excl = sums[tid] - s;
    const int len = len_fea[b];

    float run = excl;
    #pragma unroll
    for (int i = 0; i < 8; ++i) {
        run += v[i];
        const int t = tid * 8 + i;
        centers[(size_t)b * Tn + t] = run - 0.5f * v[i];
        if (t == len - 1) lens_out[b] = run;   // gls_ends[b, len-1]
    }
}

// ---------------------------------------------------------------------------
// Kernel 3: softmax weights, one block per (b,l) row.  grid = B*L, block = 256.
// Stores normalized weights as bf16 (feeds WMMA GEMM).
// ---------------------------------------------------------------------------
__global__ __launch_bounds__(256) void k_soft(const float* __restrict__ centers,
                                              const int*   __restrict__ len_fea,
                                              __bf16* __restrict__ wout)
{
    __shared__ float red[256];
    const int b   = blockIdx.x / Ln;
    const int l   = blockIdx.x % Ln;
    const int tid = threadIdx.x;
    const float pos = 0.5f + (float)l;
    const int len = len_fea[b];
    const float* cb = centers + (size_t)b * Tn;

    float sc[8];
    float mx = -3.0e38f;
    #pragma unroll
    for (int i = 0; i < 8; ++i) {
        const int t = tid * 8 + i;
        const float d = cb[t] - pos;
        float s = -(d * d) * 0.1f;            // /SIGMA2 = /10
        if (t >= len) s -= 1.0e9f;            // mask
        sc[i] = s;
        mx = fmaxf(mx, s);
    }
    red[tid] = mx;
    __syncthreads();
    for (int off = 128; off > 0; off >>= 1) {
        if (tid < off) red[tid] = fmaxf(red[tid], red[tid + off]);
        __syncthreads();
    }
    mx = red[0];
    __syncthreads();

    float sum = 0.0f;
    #pragma unroll
    for (int i = 0; i < 8; ++i) { sc[i] = __expf(sc[i] - mx); sum += sc[i]; }
    red[tid] = sum;
    __syncthreads();
    for (int off = 128; off > 0; off >>= 1) {
        if (tid < off) red[tid] += red[tid + off];
        __syncthreads();
    }
    const float inv = 1.0f / red[0];

    __bf16* wrow = wout + ((size_t)b * Ln + l) * Tn;
    #pragma unroll
    for (int i = 0; i < 8; ++i) wrow[tid * 8 + i] = (__bf16)(sc[i] * inv);
}

// ---------------------------------------------------------------------------
// Kernel 4: aligned[b,l,c] = sum_t w[b,l,t]*x[b,c,t].
// One wave per (b, m-tile of 16 L-rows, group of 4 c-tiles).  L=80=5*16, C=512.
// grid = B * 5 * 8, block = 32.  A = w (bf16, K=t contiguous per row),
// B = x (K=t contiguous per c row -> coalesced), C accum f32.
// ---------------------------------------------------------------------------
__global__ __launch_bounds__(32) void k_gemm2(const __bf16* __restrict__ wmat,
                                              const float*  __restrict__ x,
                                              float* __restrict__ out)
{
    const int id   = blockIdx.x;
    const int b    = id / 40;          // 5 m-tiles * 8 n-groups
    const int rem  = id % 40;
    const int mt   = rem / 8;
    const int ng   = rem % 8;
    const int lane = threadIdx.x & 31;
    const int tl   = lane & 15;
    const bool hi  = lane >= 16;

    const __bf16* wrow = wmat + ((size_t)b * Ln + mt * 16 + tl) * Tn;  // A row M=tl
    const float*  xb   = x + (size_t)b * Cn * Tn;

    v8f acc[4] = {};

    for (int kk = 0; kk < Tn; kk += 32) {
        // ---- A operand: w tile 16(M=l) x 32(K=t) bf16 ---------------------
        v16bf aa;
        {
            const int kb = kk + (hi ? 8 : 0);
            #pragma unroll
            for (int e = 0; e < 8; ++e) aa[e]     = wrow[kb + e];
            #pragma unroll
            for (int e = 0; e < 8; ++e) aa[8 + e] = wrow[kb + 16 + e];
        }
        // ---- 4 B operands + WMMA: x tile 32(K=t) x 16(N=c) ----------------
        #pragma unroll
        for (int j = 0; j < 4; ++j) {
            const int c = ng * 64 + j * 16 + tl;                  // N column
            const float* p = xb + (size_t)c * Tn + kk + (hi ? 8 : 0);
            v16bf bb;
            #pragma unroll
            for (int e = 0; e < 16; ++e) bb[e] = (__bf16)p[e < 8 ? e : e + 8];    // table-order K
            acc[j] = __builtin_amdgcn_wmma_f32_16x16x32_bf16(
                         false, aa, false, bb, (short)0, acc[j], false, false);
        }
    }

    // ---- store D: lane tl = N, VGPR r = M (+8 for hi half) ----------------
    #pragma unroll
    for (int j = 0; j < 4; ++j) {
        const int c = ng * 64 + j * 16 + tl;
        #pragma unroll
        for (int r = 0; r < 8; ++r) {
            const int row = mt * 16 + r + (hi ? 8 : 0);
            out[((size_t)b * Ln + row) * Cn + c] = acc[j][r];
        }
    }
}

// ---------------------------------------------------------------------------
extern "C" void kernel_launch(void* const* d_in, const int* in_sizes, int n_in,
                              void* d_out, int out_size, void* d_ws, size_t ws_size,
                              hipStream_t stream) {
    const float* x    = (const float*)d_in[0];
    const int*   lenf = (const int*)  d_in[1];
    const float* w1   = (const float*)d_in[2];
    const float* b1   = (const float*)d_in[3];
    const float* w2   = (const float*)d_in[4];
    const float* b2   = (const float*)d_in[5];
    float* out = (float*)d_out;

    // workspace layout: g[B*T] f32 | centers[B*T] f32 | w[B*L*T] bf16  (~11 MB)
    float*  g       = (float*)d_ws;
    float*  centers = g + (size_t)Bn * Tn;
    __bf16* wmat    = (__bf16*)(centers + (size_t)Bn * Tn);

    float* feat = out;                          // aligned_features [B,L,C]
    float* lens = out + (size_t)Bn * Ln * Cn;   // aligned_lengths  [B]

    k_gate <<<dim3(Bn * (Tn / 16)),       dim3(256), 0, stream>>>(x, w1, b1, w2, b2, g);
    k_scan <<<dim3(Bn),                   dim3(256), 0, stream>>>(g, lenf, centers, lens);
    k_soft <<<dim3(Bn * Ln),              dim3(256), 0, stream>>>(centers, lenf, wmat);
    k_gemm2<<<dim3(Bn * (Ln/16) * (Cn/64)), dim3(32), 0, stream>>>(wmat, x, feat);
}
